// _SparseLinear_44195213476119
// MI455X (gfx1250) — matrix operand order repaired and checked by reading, not compiled, not run on hardware
//
#include <hip/hip_runtime.h>

typedef __attribute__((ext_vector_type(16))) _Float16 v16h;
typedef __attribute__((ext_vector_type(8)))  _Float16 v8h;
typedef __attribute__((ext_vector_type(4)))  _Float16 v4h;
typedef __attribute__((ext_vector_type(8)))  float    v8f;

#define BM 128
#define BN 128
#define BK 32
#define LDSP 40   // halves per LDS row (32 data + 8 pad, keeps 16B alignment)

__device__ __forceinline__ v16h cat8(v8h a, v8h b) {
  return __builtin_shufflevector(a, b, 0,1,2,3,4,5,6,7,8,9,10,11,12,13,14,15);
}

// ---------------------------------------------------------------- zero scratch
__global__ void __launch_bounds__(256)
zero_f4(float* __restrict__ p, size_t n4) {
  size_t i      = (size_t)blockIdx.x * blockDim.x + threadIdx.x;
  size_t stride = (size_t)gridDim.x * blockDim.x;
  float4 z = make_float4(0.f, 0.f, 0.f, 0.f);
  for (; i < n4; i += stride) ((float4*)p)[i] = z;
}

// ------------------------------------------------------------- COO scatter-add
__global__ void __launch_bounds__(256)
scatter_add(const float* __restrict__ vals, const int* __restrict__ rows,
            const int* __restrict__ cols, float* __restrict__ dense,
            int nnz, int inF) {
  int i = blockIdx.x * 256 + threadIdx.x;
  if (i < nnz)
    atomicAdd(dense + (size_t)rows[i] * inF + cols[i], vals[i]);
}

// --------------------------------------- fp32 -> f16 hi/lo plane split (1 pass)
__global__ void __launch_bounds__(256)
split_f32_to_f16(const float* __restrict__ src, _Float16* __restrict__ hi,
                 _Float16* __restrict__ lo, size_t n4) {
  size_t i      = (size_t)blockIdx.x * blockDim.x + threadIdx.x;
  size_t stride = (size_t)gridDim.x * blockDim.x;
  for (; i < n4; i += stride) {
    const float4 v = ((const float4*)src)[i];
    const float f[4] = {v.x, v.y, v.z, v.w};
    v4h h, l;
    #pragma unroll
    for (int j = 0; j < 4; ++j) {
      _Float16 hv = (_Float16)f[j];
      h[j] = hv;
      l[j] = (_Float16)(f[j] - (float)hv);
    }
    ((v4h*)hi)[i] = h;
    ((v4h*)lo)[i] = l;
  }
}

// =================== FAST PATH: pre-split planes, conversion-free inner loop ==
// C[M,N] = A[M,K] * W[N,K]^T + bias ; D = Ahi*Whi + Ahi*Wlo + Alo*Whi (f32 acc)
__global__ void __launch_bounds__(256)
wmma_gemm_planes(const _Float16* __restrict__ Ahi, const _Float16* __restrict__ Alo,
                 const _Float16* __restrict__ Whi, const _Float16* __restrict__ Wlo,
                 const float* __restrict__ bias, float* __restrict__ C,
                 int M, int N, int K) {
  __shared__ _Float16 AsHi[BM * LDSP], AsLo[BM * LDSP];
  __shared__ _Float16 BsHi[BN * LDSP], BsLo[BN * LDSP];

  const int tid  = threadIdx.x;
  const int lane = tid & 31;
  const int wv   = tid >> 5;    // wave 0..7
  const int wm   = wv >> 2;     // 0..1  (64-row M slice)
  const int wn   = wv & 3;      // 0..3  (32-col N slice)
  const int hl   = lane & 15;
  const int hh   = lane >> 4;

  const int m0 = blockIdx.y * BM;
  const int n0 = blockIdx.x * BN;

  // LDS fill assignment: each thread moves one 16-byte chunk per plane
  const int lrow = tid >> 1;          // 0..127
  const int lseg = (tid & 1) * 16;    // halves 0 or 16

  v8f acc[4][2] = {};

  for (int k0 = 0; k0 < K; k0 += BK) {
    const size_t aoff = (size_t)(m0 + lrow) * K + k0 + lseg;
    const size_t woff = (size_t)(n0 + lrow) * K + k0 + lseg;
    *(v8h*)(AsHi + lrow * LDSP + lseg) = *(const v8h*)(Ahi + aoff);
    *(v8h*)(AsLo + lrow * LDSP + lseg) = *(const v8h*)(Alo + aoff);
    *(v8h*)(BsHi + lrow * LDSP + lseg) = *(const v8h*)(Whi + woff);
    *(v8h*)(BsLo + lrow * LDSP + lseg) = *(const v8h*)(Wlo + woff);
    if (k0 + BK < K) {   // pull next K tile toward L1 (global_prefetch_b8)
      __builtin_prefetch(Ahi + aoff + BK, 0, 1);
      __builtin_prefetch(Whi + woff + BK, 0, 1);
    }
    __syncthreads();

    // fragments per documented 16-bit A/B VGPR layouts
    v16h aHi[4], aLo[4], bHi[2], bLo[2];
    #pragma unroll
    for (int mt = 0; mt < 4; ++mt) {
      const int row = wm * 64 + mt * 16 + hl;
      const _Float16* ph = AsHi + row * LDSP;
      aHi[mt] = cat8(*(const v8h*)(ph + hh * 8), *(const v8h*)(ph + 16 + hh * 8));
      const _Float16* pl = AsLo + row * LDSP;
      aLo[mt] = cat8(*(const v8h*)(pl + hh * 8), *(const v8h*)(pl + 16 + hh * 8));
    }
    #pragma unroll
    for (int nt = 0; nt < 2; ++nt) {
      const int row = wn * 32 + nt * 16 + hl;
      const _Float16* ph = BsHi + row * LDSP;
      bHi[nt] = cat8(*(const v8h*)(ph + hh * 8), *(const v8h*)(ph + 16 + hh * 8));
      const _Float16* pl = BsLo + row * LDSP;
      bLo[nt] = cat8(*(const v8h*)(pl + hh * 8), *(const v8h*)(pl + 16 + hh * 8));
    }

    #pragma unroll
    for (int mt = 0; mt < 4; ++mt) {
      #pragma unroll
      for (int nt = 0; nt < 2; ++nt) {
        acc[mt][nt] = __builtin_amdgcn_wmma_f32_16x16x32_f16(
            false, aHi[mt], false, bHi[nt], (short)0, acc[mt][nt], false, false);
        acc[mt][nt] = __builtin_amdgcn_wmma_f32_16x16x32_f16(
            false, aHi[mt], false, bLo[nt], (short)0, acc[mt][nt], false, false);
        acc[mt][nt] = __builtin_amdgcn_wmma_f32_16x16x32_f16(
            false, aLo[mt], false, bHi[nt], (short)0, acc[mt][nt], false, false);
      }
    }
    __syncthreads();
  }

  #pragma unroll
  for (int nt = 0; nt < 2; ++nt) {
    const int col = n0 + wn * 32 + nt * 16 + hl;
    const float bv = bias[col];
    #pragma unroll
    for (int mt = 0; mt < 4; ++mt) {
      const int rbase = m0 + wm * 64 + mt * 16 + hh * 8;
      #pragma unroll
      for (int r = 0; r < 8; ++r)
        C[(size_t)(rbase + r) * N + col] = acc[mt][nt][r] + bv;
    }
  }
}

// ============== FALLBACK: in-kernel split (used only if ws_size is too small) ==
__global__ void __launch_bounds__(256)
wmma_gemm_convert(const float* __restrict__ A, const float* __restrict__ W,
                  const float* __restrict__ bias, float* __restrict__ C,
                  int M, int N, int K) {
  __shared__ _Float16 AsHi[BM * LDSP], AsLo[BM * LDSP];
  __shared__ _Float16 BsHi[BN * LDSP], BsLo[BN * LDSP];

  const int tid  = threadIdx.x;
  const int lane = tid & 31;
  const int wv   = tid >> 5;
  const int wm   = wv >> 2;
  const int wn   = wv & 3;
  const int hl   = lane & 15;
  const int hh   = lane >> 4;

  const int m0 = blockIdx.y * BM;
  const int n0 = blockIdx.x * BN;

  v8f acc[4][2] = {};

  for (int k0 = 0; k0 < K; k0 += BK) {
    #pragma unroll
    for (int i = 0; i < 4; ++i) {
      const int idx = tid + i * 256;
      const int row = idx >> 3;
      const int seg = (idx & 7) * 4;
      const float4 av = *(const float4*)(A + (size_t)(m0 + row) * K + k0 + seg);
      const float4 wt = *(const float4*)(W + (size_t)(n0 + row) * K + k0 + seg);
      const float af[4] = {av.x, av.y, av.z, av.w};
      const float wf[4] = {wt.x, wt.y, wt.z, wt.w};
      v4h ah, al, bh, bl;
      #pragma unroll
      for (int j = 0; j < 4; ++j) {
        _Float16 h1 = (_Float16)af[j];
        ah[j] = h1;
        al[j] = (_Float16)(af[j] - (float)h1);
        _Float16 h2 = (_Float16)wf[j];
        bh[j] = h2;
        bl[j] = (_Float16)(wf[j] - (float)h2);
      }
      *(v4h*)(AsHi + row * LDSP + seg) = ah;
      *(v4h*)(AsLo + row * LDSP + seg) = al;
      *(v4h*)(BsHi + row * LDSP + seg) = bh;
      *(v4h*)(BsLo + row * LDSP + seg) = bl;
    }
    __syncthreads();

    v16h aHi[4], aLo[4], bHi[2], bLo[2];
    #pragma unroll
    for (int mt = 0; mt < 4; ++mt) {
      const int row = wm * 64 + mt * 16 + hl;
      const _Float16* ph = AsHi + row * LDSP;
      aHi[mt] = cat8(*(const v8h*)(ph + hh * 8), *(const v8h*)(ph + 16 + hh * 8));
      const _Float16* pl = AsLo + row * LDSP;
      aLo[mt] = cat8(*(const v8h*)(pl + hh * 8), *(const v8h*)(pl + 16 + hh * 8));
    }
    #pragma unroll
    for (int nt = 0; nt < 2; ++nt) {
      const int row = wn * 32 + nt * 16 + hl;
      const _Float16* ph = BsHi + row * LDSP;
      bHi[nt] = cat8(*(const v8h*)(ph + hh * 8), *(const v8h*)(ph + 16 + hh * 8));
      const _Float16* pl = BsLo + row * LDSP;
      bLo[nt] = cat8(*(const v8h*)(pl + hh * 8), *(const v8h*)(pl + 16 + hh * 8));
    }

    #pragma unroll
    for (int mt = 0; mt < 4; ++mt) {
      #pragma unroll
      for (int nt = 0; nt < 2; ++nt) {
        acc[mt][nt] = __builtin_amdgcn_wmma_f32_16x16x32_f16(
            false, aHi[mt], false, bHi[nt], (short)0, acc[mt][nt], false, false);
        acc[mt][nt] = __builtin_amdgcn_wmma_f32_16x16x32_f16(
            false, aHi[mt], false, bLo[nt], (short)0, acc[mt][nt], false, false);
        acc[mt][nt] = __builtin_amdgcn_wmma_f32_16x16x32_f16(
            false, aLo[mt], false, bHi[nt], (short)0, acc[mt][nt], false, false);
      }
    }
    __syncthreads();
  }

  #pragma unroll
  for (int nt = 0; nt < 2; ++nt) {
    const int col = n0 + wn * 32 + nt * 16 + hl;
    const float bv = bias[col];
    #pragma unroll
    for (int mt = 0; mt < 4; ++mt) {
      const int rbase = m0 + wm * 64 + mt * 16 + hh * 8;
      #pragma unroll
      for (int r = 0; r < 8; ++r)
        C[(size_t)(rbase + r) * N + col] = acc[mt][nt][r] + bv;
    }
  }
}

// ---------------------------------------------------------------------- launch
extern "C" void kernel_launch(void* const* d_in, const int* in_sizes, int n_in,
                              void* d_out, int out_size, void* d_ws, size_t ws_size,
                              hipStream_t stream) {
  const float* vals   = (const float*)d_in[0];
  const int*   rows   = (const int*)d_in[1];
  const int*   cols   = (const int*)d_in[2];
  const float* weight = (const float*)d_in[3];
  const float* bias   = (const float*)d_in[4];

  const int nnz = in_sizes[0];
  const int OUT = in_sizes[4];              // 4096
  const int IN  = in_sizes[3] / OUT;        // 4096
  const int M   = out_size / OUT;           // 8192

  const size_t aElems = (size_t)M * IN;     // 32M
  const size_t wElems = (size_t)OUT * IN;   // 16M

  char* ws = (char*)d_ws;
  float* dense = (float*)ws;                          // aElems * 4B = 128 MB
  const size_t denseBytes = aElems * sizeof(float);

  // scratch needed for plane path: dense + Ahi/Alo + Whi/Wlo
  const size_t planeBytes = (2 * aElems + 2 * wElems) * sizeof(_Float16); // 192 MB
  const bool usePlanes = ws_size >= denseBytes + planeBytes;

  zero_f4<<<4096, 256, 0, stream>>>(dense, aElems / 4);
  scatter_add<<<(nnz + 255) / 256, 256, 0, stream>>>(vals, rows, cols, dense, nnz, IN);

  dim3 grid(OUT / BN, M / BM);              // (32, 64)

  if (usePlanes) {
    _Float16* aHi = (_Float16*)(ws + denseBytes);
    _Float16* aLo = aHi + aElems;
    _Float16* wHi = aLo + aElems;
    _Float16* wLo = wHi + wElems;
    split_f32_to_f16<<<4096, 256, 0, stream>>>(dense,  aHi, aLo, aElems / 4);
    split_f32_to_f16<<<2048, 256, 0, stream>>>(weight, wHi, wLo, wElems / 4);
    wmma_gemm_planes<<<grid, 256, 0, stream>>>(aHi, aLo, wHi, wLo, bias,
                                               (float*)d_out, M, OUT, IN);
  } else {
    wmma_gemm_convert<<<grid, 256, 0, stream>>>(dense, weight, bias,
                                                (float*)d_out, M, OUT, IN);
  }
}
